// PrototypeLayer_80049600463053
// MI455X (gfx1250) — matrix operand order repaired and checked by reading, not compiled, not run on hardware
//
#include <hip/hip_runtime.h>
#include <stdint.h>

// ---------------------------------------------------------------------------
// PrototypeLayer for MI455X (gfx1250, wave32, WMMA + async-to-LDS).
//   out[b, r] = ||x_b||^2 - 2 * <x_b, p_r> + ||p_r||^2
//   p = concat-pairs(proto) @ W_mix^T + b_mix
// bf16 WMMA (f32 accumulate) for both GEMMs; 256MB output write is the HBM
// floor (~11us @ 23.3TB/s); x_bf16 (16MB) + p_bf16 (4MB) stay L2-resident.
// dist_gemm stages K-tiles with GLOBAL_LOAD_ASYNC_TO_LDS_B128 (ASYNCcnt),
// double-buffered so the copy of tile k+1 overlaps the WMMAs of tile k.
// ---------------------------------------------------------------------------

#define BATCH 16384
#define D     512
#define TWO_D 1024
#define NPG   64
#define P     4096   // NPG*NPG

typedef __attribute__((ext_vector_type(16))) __bf16       bf16x16;
typedef __attribute__((ext_vector_type(8)))  float        f32x8;
typedef __attribute__((ext_vector_type(4)))  float        f32x4;
typedef __attribute__((ext_vector_type(4)))  unsigned int u32x4;

union Frag {
  bf16x16      v;
  u32x4        q[2];
  unsigned int u[8];
};

static __device__ __forceinline__ unsigned int bf16_rne(float f) {
  unsigned int u = __builtin_bit_cast(unsigned int, f);
  return (u + 0x7FFFu + ((u >> 16) & 1u)) >> 16;   // round-to-nearest-even
}
static __device__ __forceinline__ unsigned int pack2(float lo, float hi) {
  return bf16_rne(lo) | (bf16_rne(hi) << 16);
}

// Generic pointer into __shared__ carries the LDS byte offset in addr[31:0]
// (flat-aperture rule).
static __device__ __forceinline__ uint32_t lds_off(const void* p) {
  return (uint32_t)(size_t)p;
}

// GLOBAL_LOAD_ASYNC_TO_LDS_B128: memory -> LDS, tracked by ASYNCcnt.
// GV mode: 64-bit global address in VGPR pair, LDS byte address in VDST VGPR.
static __device__ __forceinline__ void async_copy_b128(uint32_t lds_byte_addr,
                                                       const void* gptr) {
  asm volatile("global_load_async_to_lds_b128 %0, %1, off"
               :
               : "v"(lds_byte_addr), "v"(gptr)
               : "memory");
}

static __device__ __forceinline__ void wait_async0() {
#if __has_builtin(__builtin_amdgcn_s_wait_asynccnt)
  __builtin_amdgcn_s_wait_asynccnt(0);
#else
  asm volatile("s_wait_asynccnt 0x0" ::: "memory");
#endif
}

// ---------------------------------------------------------------------------
// Kernel 1: x (f32) -> x_bf16, xx[row] = sum(x_row^2).  One wave per row.
// ---------------------------------------------------------------------------
__global__ __launch_bounds__(256) void prep_x(const float* __restrict__ x,
                                              uint16_t* __restrict__ xb,
                                              float* __restrict__ xx) {
  int row  = (int)((blockIdx.x * blockDim.x + threadIdx.x) >> 5);
  int lane = (int)(threadIdx.x & 31);
  if (row >= BATCH) return;
  const f32x4* src = (const f32x4*)(x + (size_t)row * D + lane * 16);
  f32x4 f[4];
#pragma unroll
  for (int i = 0; i < 4; ++i) f[i] = src[i];
  float s = 0.f;
  unsigned int o[8];
#pragma unroll
  for (int i = 0; i < 4; ++i) {
    s += f[i][0]*f[i][0] + f[i][1]*f[i][1] + f[i][2]*f[i][2] + f[i][3]*f[i][3];
    o[i*2+0] = pack2(f[i][0], f[i][1]);
    o[i*2+1] = pack2(f[i][2], f[i][3]);
  }
  u32x4* dst = (u32x4*)(xb + (size_t)row * D + lane * 16);
  u32x4 d0 = {o[0], o[1], o[2], o[3]};
  u32x4 d1 = {o[4], o[5], o[6], o[7]};
  dst[0] = d0;
  dst[1] = d1;
#pragma unroll
  for (int off = 16; off > 0; off >>= 1) s += __shfl_xor(s, off, 32);
  if (lane == 0) xx[row] = s;
}

// ---------------------------------------------------------------------------
// Kernel 2: mixed prototypes GEMM via WMMA bf16.
//   p[r, n] = sum_k pairs[r, k] * W_mix[n, k] + b_mix[n]
//   pairs[r] = concat(proto0[r/64], proto1[r%64])  (K split at 512, BK=32 ok)
// One 16x16 output tile per wave; 8 waves per block.
// ---------------------------------------------------------------------------
__global__ __launch_bounds__(256) void mix_protos(const float* __restrict__ protos,
                                                  const float* __restrict__ Wmix,
                                                  const float* __restrict__ bmix,
                                                  uint16_t* __restrict__ pb) {
  int w    = (int)(threadIdx.x >> 5);
  int lane = (int)(threadIdx.x & 31);
  int tile = (int)blockIdx.x * 8 + w;     // 8192 tiles total
  int tm   = tile >> 5;                   // 0..255 : 16-row tile of pairs
  int tn   = tile & 31;                   // 0..31  : 16-col tile of outputs
  int h    = lane >> 4;                   // lane half (K-block select)
  int l16  = lane & 15;

  int r = tm * 16 + l16;                  // pair row for A fragment
  int j = r >> 6, l = r & 63;
  int n = tn * 16 + l16;                  // output col for B fragment

  const float* a0   = protos + (size_t)j * D;           // proto[0][j]
  const float* a1   = protos + (size_t)(NPG + l) * D;   // proto[1][l]
  const float* brow = Wmix + (size_t)n * TWO_D;         // W_mix row n (K=2d)

  f32x8 acc = {0.f, 0.f, 0.f, 0.f, 0.f, 0.f, 0.f, 0.f};

  for (int kb = 0; kb < TWO_D / 32; ++kb) {
    int k0 = kb * 32;
    const float* asrc = (k0 < D) ? (a0 + k0) : (a1 + (k0 - D));
    Frag A, B;
    // A 16x32 bf16 layout: elems 0..7 = K(h*8 .. h*8+7), elems 8..15 = K(16+h*8 ..)
    {
      f32x4 p0 = *(const f32x4*)(asrc + h * 8);
      f32x4 p1 = *(const f32x4*)(asrc + h * 8 + 4);
      f32x4 p2 = *(const f32x4*)(asrc + 16 + h * 8);
      f32x4 p3 = *(const f32x4*)(asrc + 16 + h * 8 + 4);
      A.u[0] = pack2(p0[0], p0[1]); A.u[1] = pack2(p0[2], p0[3]);
      A.u[2] = pack2(p1[0], p1[1]); A.u[3] = pack2(p1[2], p1[3]);
      A.u[4] = pack2(p2[0], p2[1]); A.u[5] = pack2(p2[2], p2[3]);
      A.u[6] = pack2(p3[0], p3[1]); A.u[7] = pack2(p3[2], p3[3]);
    }
    // B 32x16 bf16 layout: elems 0..7 = K(h*8 .. h*8+7), elems 8..15 = K(16+h*8 ..)
    {
      const float* bsrc = brow + k0 + h * 8;
#pragma unroll
      for (int i = 0; i < 4; ++i) {
        f32x4 pk = *(const f32x4*)(bsrc + i * 4 + (i >> 1) * 8);
        B.u[i*2+0] = pack2(pk[0], pk[1]);
        B.u[i*2+1] = pack2(pk[2], pk[3]);
      }
    }
    acc = __builtin_amdgcn_wmma_f32_16x16x32_bf16(
        false, A.v, false, B.v, (short)0, acc, false, false);
  }

  float bias = bmix[n];
  // C layout: VGPR g -> row (g + 8*h), col = l16
#pragma unroll
  for (int g = 0; g < 8; ++g) {
    int row = tm * 16 + g + 8 * h;
    pb[(size_t)row * D + n] = (uint16_t)bf16_rne(acc[g] + bias);
  }
}

// ---------------------------------------------------------------------------
// Kernel 3: pp[r] = sum over bf16 p row of squares (consistent with GEMM).
// ---------------------------------------------------------------------------
__global__ __launch_bounds__(256) void pp_kernel(const uint16_t* __restrict__ pb,
                                                 float* __restrict__ pp) {
  int r = (int)(blockIdx.x * blockDim.x + threadIdx.x);
  if (r >= P) return;
  const u32x4* src = (const u32x4*)(pb + (size_t)r * D);
  float s = 0.f;
  for (int i = 0; i < D / 8; ++i) {
    u32x4 q = src[i];
#pragma unroll
    for (int jj = 0; jj < 4; ++jj) {
      float lo = __builtin_bit_cast(float, q[jj] << 16);
      float hi = __builtin_bit_cast(float, q[jj] & 0xFFFF0000u);
      s += lo * lo + hi * hi;
    }
  }
  pp[r] = s;
}

// ---------------------------------------------------------------------------
// Kernel 4: main distance GEMM, bf16 WMMA, async double-buffered LDS tiles.
//   Block tile 128x128, BK=32, 8 waves as 4x2 grid; wave tile 32x64
//   (2 M-tiles x 4 N-tiles = 8 f32x8 accumulators).
//   LDS rows padded to 40 bf16 (80B) -> conflict-free b128 fragment loads.
//   Tile k+1 staged via GLOBAL_LOAD_ASYNC_TO_LDS_B128 while tile k computes.
// ---------------------------------------------------------------------------
#define BM   128
#define BN   128
#define BK   32
#define LSTR 40
#define KSTEPS (D / BK)

__global__ __launch_bounds__(256) void dist_gemm(const uint16_t* __restrict__ xb,
                                                 const uint16_t* __restrict__ pb,
                                                 const float* __restrict__ xx,
                                                 const float* __restrict__ pp,
                                                 float* __restrict__ out) {
  __shared__ uint16_t sA[2][BM * LSTR];   // 2 x 10240 B
  __shared__ uint16_t sB[2][BN * LSTR];   // 2 x 10240 B

  int tid  = (int)threadIdx.x;
  int bn   = (int)(blockIdx.x & 31);   // 32 N-blocks
  int bm   = (int)(blockIdx.x >> 5);   // 128 M-blocks
  int w    = tid >> 5;
  int lane = tid & 31;
  int wm   = w >> 1;                   // 0..3
  int wn   = w & 1;                    // 0..1
  int h    = lane >> 4;
  int l16  = lane & 15;

  // Cooperative tile loader: thread t covers 16 bf16 (32B) of one row,
  // i.e. two async b128 transfers per operand per K-step.
  int ldr = tid >> 1;                  // 0..127
  int ldc = (tid & 1) * 16;            // 0 or 16
  const uint16_t* gArow = xb + (size_t)(bm * BM + ldr) * D + ldc;
  const uint16_t* gBrow = pb + (size_t)(bn * BN + ldr) * D + ldc;
  uint32_t laoff[2] = { lds_off(&sA[0][ldr * LSTR + ldc]),
                        lds_off(&sA[1][ldr * LSTR + ldc]) };
  uint32_t lboff[2] = { lds_off(&sB[0][ldr * LSTR + ldc]),
                        lds_off(&sB[1][ldr * LSTR + ldc]) };

  f32x8 acc[2][4];
#pragma unroll
  for (int mt = 0; mt < 2; ++mt)
#pragma unroll
    for (int nt = 0; nt < 4; ++nt)
      acc[mt][nt] = (f32x8){0.f, 0.f, 0.f, 0.f, 0.f, 0.f, 0.f, 0.f};

  // Prologue: stage tile 0 into buffer 0.
  async_copy_b128(laoff[0],      gArow);
  async_copy_b128(laoff[0] + 16, gArow + 8);
  async_copy_b128(lboff[0],      gBrow);
  async_copy_b128(lboff[0] + 16, gBrow + 8);

  for (int kb = 0; kb < KSTEPS; ++kb) {
    int cur = kb & 1;
    wait_async0();        // my async copies into buffer `cur` landed
    __syncthreads();      // everyone's copies landed; prev compute finished

    if (kb + 1 < KSTEPS) {  // prefetch tile kb+1 into the other buffer
      int nxt = cur ^ 1;
      const uint16_t* ga = gArow + (kb + 1) * BK;
      const uint16_t* gb = gBrow + (kb + 1) * BK;
      async_copy_b128(laoff[nxt],      ga);
      async_copy_b128(laoff[nxt] + 16, ga + 8);
      async_copy_b128(lboff[nxt],      gb);
      async_copy_b128(lboff[nxt] + 16, gb + 8);
    }

    Frag Af[2], Bf[4];
#pragma unroll
    for (int mt = 0; mt < 2; ++mt) {
      const uint16_t* base = &sA[cur][(wm * 32 + mt * 16 + l16) * LSTR];
      Af[mt].q[0] = *(const u32x4*)(base + h * 8);        // K 0..7 / 8..15
      Af[mt].q[1] = *(const u32x4*)(base + 16 + h * 8);   // K 16..23 / 24..31
    }
#pragma unroll
    for (int nt = 0; nt < 4; ++nt) {
      const uint16_t* base = &sB[cur][(wn * 64 + nt * 16 + l16) * LSTR];
      Bf[nt].q[0] = *(const u32x4*)(base + h * 8);        // K 0..7 / 8..15
      Bf[nt].q[1] = *(const u32x4*)(base + 16 + h * 8);   // K 16..23 / 24..31
    }
#pragma unroll
    for (int mt = 0; mt < 2; ++mt)
#pragma unroll
      for (int nt = 0; nt < 4; ++nt)
        acc[mt][nt] = __builtin_amdgcn_wmma_f32_16x16x32_bf16(
            false, Af[mt].v, false, Bf[nt].v, (short)0, acc[mt][nt], false, false);
    // No trailing barrier: next iteration's wait+barrier protects buffer reuse,
    // and the prefetch targets the opposite buffer.
  }

  // Epilogue: out = xx[row] + pp[col] - 2*dot
  int colb = bn * BN + wn * 64;
  float ppv[4];
#pragma unroll
  for (int nt = 0; nt < 4; ++nt) ppv[nt] = pp[colb + nt * 16 + l16];

#pragma unroll
  for (int mt = 0; mt < 2; ++mt) {
    int rowb = bm * BM + wm * 32 + mt * 16 + 8 * h;
#pragma unroll
    for (int g = 0; g < 8; ++g) {
      int row  = rowb + g;
      float xr = xx[row];
      float* o = out + (size_t)row * P + colb + l16;
#pragma unroll
      for (int nt = 0; nt < 4; ++nt)
        o[nt * 16] = __builtin_fmaf(-2.0f, acc[mt][nt][g], xr + ppv[nt]);
    }
  }
}

// ---------------------------------------------------------------------------
// Launch. Workspace layout (~21 MB):
//   [0)                      x_bf16  : 16384*512*2 = 16 MiB
//   [+16MiB)                 p_bf16  :  4096*512*2 =  4 MiB
//   [+20MiB)                 xx      : 16384*4
//   [+20MiB+64KiB)           pp      :  4096*4
// ---------------------------------------------------------------------------
extern "C" void kernel_launch(void* const* d_in, const int* in_sizes, int n_in,
                              void* d_out, int out_size, void* d_ws, size_t ws_size,
                              hipStream_t stream) {
  const float* x      = (const float*)d_in[0];
  const float* protos = (const float*)d_in[1];
  const float* Wmix   = (const float*)d_in[2];
  const float* bmix   = (const float*)d_in[3];
  float* out          = (float*)d_out;

  char* ws = (char*)d_ws;
  uint16_t* xb = (uint16_t*)ws;
  uint16_t* pb = (uint16_t*)(ws + (size_t)BATCH * D * 2);
  float*    xx = (float*)(ws + (size_t)BATCH * D * 2 + (size_t)P * D * 2);
  float*    pp = (float*)((char*)xx + (size_t)BATCH * 4);

  prep_x    <<<BATCH / 8,                 256, 0, stream>>>(x, xb, xx);
  mix_protos<<<(P / 16) * (D / 16) / 8,   256, 0, stream>>>(protos, Wmix, bmix, pb);
  pp_kernel <<<P / 256,                   256, 0, stream>>>(pb, pp);
  dist_gemm <<<(BATCH / BM) * (P / BN),   256, 0, stream>>>(xb, pb, xx, pp, out);
}
